// CAREConv_24764781429368
// MI455X (gfx1250) — matrix operand order repaired and checked by reading, not compiled, not run on hardware
//
#include <hip/hip_runtime.h>
#include <hip/hip_bf16.h>
#include <math.h>

// ---------------------------------------------------------------------------
// CAREConv fused pipeline for MI455X (gfx1250, wave32).
//   K1: mlp_h = tanh(feat @ W_mlp + b_mlp)          (one wave / node, f32)
//   K2: per-etype L1-dist -> rank-select 16/32 -> gathered mean -> tanh,
//       combine 3 etypes + residual + tanh, store activations as f16.
//   K3: pack W_lin into per-lane WMMA-B layout (f16).
//   K4: out = h16 @ W16 + b_lin via v_wmma_f32_16x16x32_f16 (f32 accum).
// ---------------------------------------------------------------------------

typedef __attribute__((ext_vector_type(16))) _Float16 v16h;
typedef __attribute__((ext_vector_type(8)))  _Float16 v8h;
typedef __attribute__((ext_vector_type(4)))  _Float16 v4h;
typedef __attribute__((ext_vector_type(8)))  float    v8f;

#define D_IN  128
#define D_OUT 64
#define KNBR  32
#define NSEL  16   // ceil(32 * 0.5)

// ---- K1: per-node 128->2 MLP + tanh; one wave32 per node -------------------
__global__ void __launch_bounds__(256)
mlp_tanh_kernel(const float* __restrict__ feat, const float* __restrict__ Wm,
                const float* __restrict__ bm, float* __restrict__ mlp_h, int N) {
  const int n    = blockIdx.x * (blockDim.x >> 5) + (threadIdx.x >> 5);
  const int lane = threadIdx.x & 31;
  if (n >= N) return;
  const float4 x = *(const float4*)(feat + (size_t)n * D_IN + lane * 4);
  const int k = lane * 4;
  float a0 = x.x * Wm[(k + 0) * 2 + 0] + x.y * Wm[(k + 1) * 2 + 0] +
             x.z * Wm[(k + 2) * 2 + 0] + x.w * Wm[(k + 3) * 2 + 0];
  float a1 = x.x * Wm[(k + 0) * 2 + 1] + x.y * Wm[(k + 1) * 2 + 1] +
             x.z * Wm[(k + 2) * 2 + 1] + x.w * Wm[(k + 3) * 2 + 1];
#pragma unroll
  for (int off = 16; off; off >>= 1) {
    a0 += __shfl_xor(a0, off);
    a1 += __shfl_xor(a1, off);
  }
  if (lane == 0) {
    mlp_h[2 * (size_t)n + 0] = tanhf(a0 + bm[0]);
    mlp_h[2 * (size_t)n + 1] = tanhf(a1 + bm[1]);
  }
}

// ---- K2: distance + top-16-smallest select + gathered mean, all 3 etypes ---
// One wave32 per node. lane k owns edge k; rank-select in registers; the
// gather of each selected 512B feat row is a fully coalesced wave read.
__global__ void __launch_bounds__(256)
select_aggregate_kernel(const float* __restrict__ feat,
                        const float* __restrict__ mlp_h,
                        const int* __restrict__ nbr0,
                        const int* __restrict__ nbr1,
                        const int* __restrict__ nbr2,
                        _Float16* __restrict__ h16, int N) {
  const int n    = blockIdx.x * (blockDim.x >> 5) + (threadIdx.x >> 5);
  const int lane = threadIdx.x & 31;
  if (n >= N) return;

  const float mh0 = mlp_h[2 * (size_t)n + 0];
  const float mh1 = mlp_h[2 * (size_t)n + 1];

  float hs0 = 0.f, hs1 = 0.f, hs2 = 0.f, hs3 = 0.f;  // 0.5 * sum tanh(mean)
  const int* const nbrs[3] = {nbr0, nbr1, nbr2};

#pragma unroll
  for (int e = 0; e < 3; ++e) {
    const int src = nbrs[e][(size_t)n * KNBR + lane];
    const float m0 = mlp_h[2 * (size_t)src + 0];
    const float m1 = mlp_h[2 * (size_t)src + 1];
    const float d  = fabsf(m0 - mh0) + fabsf(m1 - mh1);

    // Stable rank: # of edges strictly smaller, or equal with lower index.
    int rank = 0;
#pragma unroll
    for (int j = 0; j < 32; ++j) {
      const float dj = __shfl(d, j);
      rank += (dj < d) || (dj == d && j < lane);
    }
    unsigned long long mask = __ballot(rank < NSEL);  // exactly 16 bits set

    float a0 = 0.f, a1 = 0.f, a2 = 0.f, a3 = 0.f;
    for (int s = 0; s < NSEL; ++s) {
      const int b = __ffsll(mask) - 1;
      mask &= (mask - 1ull);
      const int srow = __shfl(src, b);
      if (mask) {  // wave-uniform: prefetch next selected row into L2/L0
        const int bn   = __ffsll(mask) - 1;
        const int srn  = __shfl(src, bn);
        __builtin_prefetch(feat + (size_t)srn * D_IN + lane * 4, 0, 1);
      }
      const float4 f = *(const float4*)(feat + (size_t)srow * D_IN + lane * 4);
      a0 += f.x; a1 += f.y; a2 += f.z; a3 += f.w;
    }
    const float inv = 1.0f / NSEL;
    hs0 += 0.5f * tanhf(a0 * inv);
    hs1 += 0.5f * tanhf(a1 * inv);
    hs2 += 0.5f * tanhf(a2 * inv);
    hs3 += 0.5f * tanhf(a3 * inv);
  }

  const float4 fn = *(const float4*)(feat + (size_t)n * D_IN + lane * 4);
  v4h o;
  o[0] = (_Float16)tanhf(fn.x + hs0);
  o[1] = (_Float16)tanhf(fn.y + hs1);
  o[2] = (_Float16)tanhf(fn.z + hs2);
  o[3] = (_Float16)tanhf(fn.w + hs3);
  *(v4h*)(h16 + (size_t)n * D_IN + lane * 4) = o;
}

// ---- K3: pack W_lin [128,64] f32 -> f16 in per-lane WMMA-B layout ----------
// B16[((ntile*4 + kk)*32 + lane)*16 + i] = W[(32*kk + (lane>>4)*16 + i), ntile*16 + (lane&15)]
__global__ void __launch_bounds__(256)
pack_w_kernel(const float* __restrict__ Wlin, _Float16* __restrict__ B16) {
  const int idx = blockIdx.x * blockDim.x + threadIdx.x;
  if (idx >= 4 * 4 * 32 * 16) return;
  const int i     = idx & 15;
  const int lane  = (idx >> 4) & 31;
  const int kk    = (idx >> 9) & 3;
  const int ntile = idx >> 11;
  const int k     = kk * 32 + (lane >> 4) * 16 + i;
  const int ncol  = ntile * 16 + (lane & 15);
  B16[idx] = (_Float16)Wlin[k * D_OUT + ncol];
}

// ---- K4: out[N,64] = h16[N,128] @ W16 + b_lin via WMMA ---------------------
// 128-thread block = 4 waves; block owns 16 rows; wave w owns N-tile w.
__global__ void __launch_bounds__(128)
gemm_wmma_kernel(const _Float16* __restrict__ h16,
                 const _Float16* __restrict__ B16,
                 const float* __restrict__ bl,
                 float* __restrict__ out, int N) {
  const int lane = threadIdx.x & 31;
  const int wv   = threadIdx.x >> 5;  // N-tile 0..3
  const int row  = lane & 15;
  const int hi   = lane >> 4;
  const int m0   = blockIdx.x * 16;

  v8f c = {};
#pragma unroll
  for (int kk = 0; kk < 4; ++kk) {
    const int kbase = kk * 32;
    int rnode = m0 + row;
    if (rnode >= N) rnode = N - 1;  // branchless clamp: EXEC stays all-ones
    // A 16x32 f16 layout: lanes 0-15 hold K=0..15, lanes 16-31 hold K=16..31.
    const _Float16* ap = h16 + (size_t)rnode * D_IN + kbase + hi * 16;
    const v8h alo = *(const v8h*)(ap);
    const v8h ahi = *(const v8h*)(ap + 8);
    v16h a;
#pragma unroll
    for (int i = 0; i < 8; ++i) { a[i] = alo[i]; a[8 + i] = ahi[i]; }
    // B pre-packed: one contiguous 32B load per lane per K-step.
    const v16h b = *(const v16h*)(B16 + ((size_t)(wv * 4 + kk) * 32 + lane) * 16);
    c = __builtin_amdgcn_wmma_f32_16x16x32_f16(
        /*neg_a=*/false, a, /*neg_b=*/false, b,
        /*c_mod=*/(short)0, c, /*reuse_a=*/false, /*reuse_b=*/false);
  }

  const int ncol  = wv * 16 + (lane & 15);
  const float bias = bl[ncol];
#pragma unroll
  for (int r = 0; r < 8; ++r) {   // D: VGPR r, lane -> (M = r + 8*hi, N = lane&15)
    const int m = m0 + r + 8 * hi;
    if (m < N) out[(size_t)m * D_OUT + ncol] = c[r] + bias;
  }
}

// ---------------------------------------------------------------------------
extern "C" void kernel_launch(void* const* d_in, const int* in_sizes, int n_in,
                              void* d_out, int out_size, void* d_ws, size_t ws_size,
                              hipStream_t stream) {
  const float* feat = (const float*)d_in[0];
  const float* Wm   = (const float*)d_in[1];
  const float* bm   = (const float*)d_in[2];
  const float* Wl   = (const float*)d_in[3];
  const float* bl   = (const float*)d_in[4];
  const int*   nbr0 = (const int*)d_in[5];
  const int*   nbr1 = (const int*)d_in[6];
  const int*   nbr2 = (const int*)d_in[7];

  const int N = in_sizes[0] / D_IN;

  // Workspace: mlp_h [N,2] f32 | h16 [N,128] f16 | B16 [4*4*32*16] f16
  char* ws = (char*)d_ws;
  float* mlp_h = (float*)ws;
  size_t off = ((size_t)N * 2 * sizeof(float) + 255) & ~(size_t)255;
  _Float16* h16 = (_Float16*)(ws + off);
  off += (size_t)N * D_IN * sizeof(_Float16);      // N*256 bytes (256-aligned)
  _Float16* B16 = (_Float16*)(ws + off);

  const int wavesPerBlock = 8;  // 256 threads
  const int nodeBlocks = (N + wavesPerBlock - 1) / wavesPerBlock;

  mlp_tanh_kernel<<<nodeBlocks, 256, 0, stream>>>(feat, Wm, bm, mlp_h, N);
  pack_w_kernel<<<32, 256, 0, stream>>>(Wl, B16);
  select_aggregate_kernel<<<nodeBlocks, 256, 0, stream>>>(
      feat, mlp_h, nbr0, nbr1, nbr2, h16, N);
  gemm_wmma_kernel<<<(N + 15) / 16, 128, 0, stream>>>(
      h16, B16, bl, (float*)d_out, N);
}
